// Attention_35296041239157
// MI455X (gfx1250) — matrix operand order repaired and checked by reading, not compiled, not run on hardware
//
#include <hip/hip_runtime.h>
#include <hip/hip_bf16.h>
#include <math.h>

typedef __attribute__((ext_vector_type(16))) __bf16 v16bf;
typedef __attribute__((ext_vector_type(8)))  float  v8f;

#define BATCH 16
#define SEQL  4096
#define DIM   1024
#define HID   512
#define TM    64          // rows per workgroup in score kernel
#define MT    4           // 16-row M tiles per workgroup
#define KSTEPS 32         // 1024 / 32
#define NTILES 64         // 1024 / 16

// ---------------------------------------------------------------------------
// Kernel 0: convert Wh (f32, row-major [k][n]) into bf16 pre-swizzled into the
// contiguous-k WMMA B-fragment layout used by this file (32x16 bf16 tile -> 16 bf16 per lane):
//   frag index = ((nt*32 + ks)*32 + lane), element j:
//     n = nt*16 + lane%16 ;  k = ks*32 + (lane/16)*16 + j
// ---------------------------------------------------------------------------
__global__ __launch_bounds__(256) void wh_swizzle_kernel(
    const float* __restrict__ Wh, __bf16* __restrict__ whbf)
{
    int idx = blockIdx.x * 256 + threadIdx.x;     // 64*32*32*16 = 1,048,576
    int j  = idx & 15;
    int ln = (idx >> 4) & 31;
    int ks = (idx >> 9) & 31;
    int nt = idx >> 14;
    int n = nt * 16 + (ln & 15);
    int k = ks * 32 + (ln >> 4) * 16 + j;
    whbf[idx] = (__bf16)Wh[(size_t)k * DIM + n];
}

// ---------------------------------------------------------------------------
// Kernel 1: dec_feat[b][e] = sum_d s_t[b][d] * Ws_w[d][e] + Ws_b[e]
//           s_t = concat(h_dec, c_dec)   (16 x 1024 output, tiny GEMV)
// ---------------------------------------------------------------------------
__global__ __launch_bounds__(256) void dec_feat_kernel(
    const float* __restrict__ h_dec, const float* __restrict__ c_dec,
    const float* __restrict__ Ws_w,  const float* __restrict__ Ws_b,
    float* __restrict__ dec_feat)
{
    __shared__ float s[DIM];
    const int b  = blockIdx.y;
    const int e0 = blockIdx.x * 256;
    const int tid = threadIdx.x;
    for (int i = tid; i < DIM; i += 256)
        s[i] = (i < HID) ? h_dec[b * HID + i] : c_dec[b * HID + (i - HID)];
    __syncthreads();
    const int e = e0 + tid;
    float acc = Ws_b[e];
    for (int d = 0; d < DIM; ++d)
        acc += s[d] * Ws_w[(size_t)d * DIM + e];
    dec_feat[b * DIM + e] = acc;
}

// ---------------------------------------------------------------------------
// Kernel 2: fused  score[b][l] = v . tanh(enc[b][l] @ Wh + dec_feat[b] + cov*wc)
// One WG = 64 rows of one batch. enc tile (64x1024) -> bf16 A-fragment layout
// in 128 KB dynamic LDS. 8 waves x 8 N-tiles; B frag loaded once per (nt,ks),
// reused across 4 M-tiles. 8192 v_wmma_f32_16x16x32_bf16 per WG.
// A-fragment layout (16-bit A 16x32, contiguous k as in B, not ISA order): lane l (h=l/16, m=l%16),
//   element j: k_local = h*16+j
// ---------------------------------------------------------------------------
__global__ __launch_bounds__(256) void score_kernel(
    const float*  __restrict__ enc,  const float* __restrict__ cov,
    const float*  __restrict__ wc,   const float* __restrict__ v,
    const float*  __restrict__ dec_feat, const __bf16* __restrict__ whbf,
    float* __restrict__ score)
{
    extern __shared__ __align__(32) __bf16 a_lds[];   // [MT][KSTEPS][32][16]
    __shared__ float s_score[TM];
    __shared__ float s_cov[TM];

    const int b    = blockIdx.y;
    const int l0   = blockIdx.x * TM;
    const int tid  = threadIdx.x;
    const int lane = tid & 31;
    const int wave = tid >> 5;

    if (tid < TM) { s_cov[tid] = cov[b * SEQL + l0 + tid]; s_score[tid] = 0.0f; }

    // Load enc tile, convert f32->bf16, store in A-fragment order.
    const float* encb = enc + ((size_t)(b * SEQL + l0)) * DIM;
    for (int idx = tid; idx < MT * KSTEPS * 32 * 16; idx += 256) {
        int j  = idx & 15;
        int ln = (idx >> 4) & 31;
        int ks = (idx >> 9) & 31;
        int mt = idx >> 14;
        int h = ln >> 4, m = ln & 15;
        int kl = h * 16 + j;
        int k  = ks * 32 + kl;
        a_lds[idx] = (__bf16)encb[(size_t)(mt * 16 + m) * DIM + k];
    }
    __syncthreads();

    float sacc[MT][8];
#pragma unroll
    for (int mt = 0; mt < MT; ++mt)
#pragma unroll
        for (int r = 0; r < 8; ++r) sacc[mt][r] = 0.0f;

    const v16bf* aWfrag = (const v16bf*)a_lds;

    for (int t = 0; t < NTILES / 8; ++t) {
        const int nt = wave * (NTILES / 8) + t;          // this wave's N tile
        const int e  = nt * 16 + (lane & 15);            // output column of lane

        v8f acc[MT];
#pragma unroll
        for (int mt = 0; mt < MT; ++mt) acc[mt] = (v8f)(0.0f);

        const v16bf* bfrag_base = (const v16bf*)whbf + (size_t)nt * KSTEPS * 32 + lane;
#pragma unroll 4
        for (int ks = 0; ks < KSTEPS; ++ks) {
            v16bf bfrag = bfrag_base[ks * 32];
#pragma unroll
            for (int mt = 0; mt < MT; ++mt) {
                v16bf afrag = aWfrag[(mt * KSTEPS + ks) * 32 + lane];
                acc[mt] = __builtin_amdgcn_wmma_f32_16x16x32_bf16(
                    false, afrag, false, bfrag, (short)0, acc[mt], false, false);
            }
        }

        const float wce = wc[e];
        const float ve  = v[e];
        const float dfe = dec_feat[b * DIM + e];
#pragma unroll
        for (int mt = 0; mt < MT; ++mt) {
#pragma unroll
            for (int r = 0; r < 8; ++r) {
                int m = mt * 16 + r + 8 * (lane >> 4);   // row within the 64
                float val = acc[mt][r] + dfe + s_cov[m] * wce;
                sacc[mt][r] += tanhf(val) * ve;
            }
        }
    }

#pragma unroll
    for (int mt = 0; mt < MT; ++mt)
#pragma unroll
        for (int r = 0; r < 8; ++r) {
            int m = mt * 16 + r + 8 * (lane >> 4);
            atomicAdd(&s_score[m], sacc[mt][r]);         // ds_add_f32
        }
    __syncthreads();
    if (tid < TM) score[b * SEQL + l0 + tid] = s_score[tid];
}

// ---------------------------------------------------------------------------
// Kernel 3: per-batch masked softmax + coverage update.
// attn = (exp(s-max)*mask) / sum(exp(s-max)*mask)   (== reference two-step)
// ---------------------------------------------------------------------------
__global__ __launch_bounds__(256) void softmax_kernel(
    const float* __restrict__ score, const float* __restrict__ mask,
    const float* __restrict__ cov,
    float* __restrict__ attn_out, float* __restrict__ covnew_out)
{
    __shared__ float red[256];
    const int b   = blockIdx.x;
    const int tid = threadIdx.x;
    const float* sb = score + b * SEQL;
    const float* mb = mask  + b * SEQL;

    float mx = -INFINITY;
    for (int l = tid; l < SEQL; l += 256) mx = fmaxf(mx, sb[l]);
    red[tid] = mx; __syncthreads();
    for (int s = 128; s > 0; s >>= 1) {
        if (tid < s) red[tid] = fmaxf(red[tid], red[tid + s]);
        __syncthreads();
    }
    mx = red[0]; __syncthreads();

    float msum = 0.0f;
    for (int l = tid; l < SEQL; l += 256) msum += __expf(sb[l] - mx) * mb[l];
    red[tid] = msum; __syncthreads();
    for (int s = 128; s > 0; s >>= 1) {
        if (tid < s) red[tid] += red[tid + s];
        __syncthreads();
    }
    const float inv = 1.0f / red[0];

    for (int l = tid; l < SEQL; l += 256) {
        float a = __expf(sb[l] - mx) * mb[l] * inv;
        attn_out[b * SEQL + l]   = a;
        covnew_out[b * SEQL + l] = cov[b * SEQL + l] + a;
    }
}

// ---------------------------------------------------------------------------
// Kernel 4: context[b][d] = sum_l attn[b][l] * enc[b][l][d]  (memory bound)
// ---------------------------------------------------------------------------
__global__ __launch_bounds__(256) void zero_kernel(float* __restrict__ p, int n)
{
    int i = blockIdx.x * 256 + threadIdx.x;
    if (i < n) p[i] = 0.0f;
}

__global__ __launch_bounds__(256) void context_kernel(
    const float* __restrict__ enc, const float* __restrict__ attn,
    float* __restrict__ ctx)
{
    const int b   = blockIdx.y;
    const int lc  = blockIdx.x;                 // 32 chunks of 128 rows
    const int tid = threadIdx.x;
    float acc0 = 0.f, acc1 = 0.f, acc2 = 0.f, acc3 = 0.f;
    const int lbeg = lc * (SEQL / 32), lend = lbeg + (SEQL / 32);
    for (int l = lbeg; l < lend; ++l) {
        float a = attn[b * SEQL + l];
        const float4 ev = *(const float4*)(enc + ((size_t)(b * SEQL + l)) * DIM + tid * 4);
        acc0 += a * ev.x; acc1 += a * ev.y; acc2 += a * ev.z; acc3 += a * ev.w;
    }
    atomicAdd(&ctx[b * DIM + tid * 4 + 0], acc0);
    atomicAdd(&ctx[b * DIM + tid * 4 + 1], acc1);
    atomicAdd(&ctx[b * DIM + tid * 4 + 2], acc2);
    atomicAdd(&ctx[b * DIM + tid * 4 + 3], acc3);
}

// ---------------------------------------------------------------------------
extern "C" void kernel_launch(void* const* d_in, const int* in_sizes, int n_in,
                              void* d_out, int out_size, void* d_ws, size_t ws_size,
                              hipStream_t stream)
{
    const float* enc   = (const float*)d_in[0];   // (16,4096,1024)
    const float* h_dec = (const float*)d_in[1];   // (1,16,512)
    const float* c_dec = (const float*)d_in[2];   // (1,16,512)
    const float* mask  = (const float*)d_in[3];   // (16,4096)
    const float* cov   = (const float*)d_in[4];   // (16,4096)
    const float* Wh    = (const float*)d_in[5];   // (1024,1024)
    const float* Ws_w  = (const float*)d_in[6];   // (1024,1024)
    const float* Ws_b  = (const float*)d_in[7];   // (1024,)
    const float* wc    = (const float*)d_in[8];   // (1024,)
    const float* v     = (const float*)d_in[9];   // (1024,)

    float* out     = (float*)d_out;
    float* ctx     = out;                              // (16,1024)
    float* attn    = out + BATCH * DIM;                // (16,4096)
    float* covnew  = attn + BATCH * SEQL;              // (16,4096)

    // workspace layout
    char* ws = (char*)d_ws;
    __bf16* whbf    = (__bf16*)ws;                                   // 2 MB
    float*  decfeat = (float*)(ws + (size_t)2 * 1024 * 1024);        // 64 KB
    float*  score   = (float*)(ws + (size_t)2 * 1024 * 1024 + 65536);// 256 KB

    // 0) Wh -> bf16 swizzled B fragments
    wh_swizzle_kernel<<<dim3(4096), dim3(256), 0, stream>>>(Wh, whbf);

    // 1) dec_feat
    dec_feat_kernel<<<dim3(4, BATCH), dim3(256), 0, stream>>>(
        h_dec, c_dec, Ws_w, Ws_b, decfeat);

    // 2) fused WMMA score
    const size_t a_lds_bytes = (size_t)MT * KSTEPS * 32 * 16 * sizeof(__bf16); // 128 KB
    score_kernel<<<dim3(SEQL / TM, BATCH), dim3(256), a_lds_bytes, stream>>>(
        enc, cov, wc, v, decfeat, whbf, score);

    // 3) softmax + coverage
    softmax_kernel<<<dim3(BATCH), dim3(256), 0, stream>>>(
        score, mask, cov, attn, covnew);

    // 4) context
    zero_kernel<<<dim3((BATCH * DIM + 255) / 256), dim3(256), 0, stream>>>(
        ctx, BATCH * DIM);
    context_kernel<<<dim3(32, BATCH), dim3(256), 0, stream>>>(enc, attn, ctx);
}
